// SingleLayerSNNModel_21741124453044
// MI455X (gfx1250) — matrix operand order repaired and checked by reading, not compiled, not run on hardware
//
#include <hip/hip_runtime.h>
#include <hip/hip_bf16.h>

#define ALPHA 0.9f
#define V_TH  1.0f
#define BB 32
#define TT 256
#define NN 4096
#define RR 32
#define KK 64
#define KC 64                 // K elements staged per TDM chunk
#define NCH (NN / KC)         // 64 chunks per timestep

typedef __bf16 bf16_t;
typedef __attribute__((ext_vector_type(16))) __bf16 v16bf;
typedef __attribute__((ext_vector_type(8)))  float  v8f;
typedef __attribute__((ext_vector_type(4)))  unsigned int u32x4;
typedef __attribute__((ext_vector_type(4)))  int i32x4;
typedef __attribute__((ext_vector_type(8)))  int i32x8;

union ALoad { uint4 u[2]; v16bf v; };

// ---------------------------------------------------------------------------
// Tensor Data Mover: async 2D tile load global->LDS (bf16 elements).
// D# packing per CDNA5 ISA:
//   g0: count=1 | lds_addr | global_addr[56:0] | type=2
//   g1: data_size=1(2B); tensor_dim0=NN; tensor_dim1=rows; tile_dim0=KC;
//       tile_dim1=rows; tensor_dim0_stride=NN
// Tracked with TENSORcnt; EXEC ignored (issue from one wave only).
// ---------------------------------------------------------------------------
__device__ __forceinline__ void tdm_load_tile_2d(const bf16_t* gptr,
                                                 void* lptr, int tile_rows) {
  const unsigned long long ga = (unsigned long long)(uintptr_t)gptr;
  u32x4 g0;
  g0[0] = 1u;                                           // count=1, user D#
  g0[1] = (unsigned int)(uintptr_t)lptr;                // LDS byte address
  g0[2] = (unsigned int)ga;                             // global_addr[31:0]
  g0[3] = ((unsigned int)(ga >> 32) & 0x01FFFFFFu) | 0x80000000u; // [56:32]|type=2
  i32x8 g1;
  g1[0] = 0x00010000;                                   // data_size=2 bytes
  g1[1] = (int)((NN & 0xFFFF) << 16);                   // tensor_dim0 lo16
  g1[2] = (int)(((NN >> 16) & 0xFFFF) | ((tile_rows & 0xFFFF) << 16)); // dim0 hi|dim1 lo
  g1[3] = (int)(KC << 16);                              // dim1 hi=0 | tile_dim0=KC
  g1[4] = tile_rows;                                    // tile_dim1 (tile_dim2=0)
  g1[5] = NN;                                           // tensor_dim0_stride lo32
  g1[6] = 0;                                            // stride hi, dim1_stride lo
  g1[7] = 0;
  const i32x4 z4 = {0, 0, 0, 0};
#if defined(__clang_major__) && (__clang_major__ >= 23)
  const i32x8 z8 = {0, 0, 0, 0, 0, 0, 0, 0};
  __builtin_amdgcn_tensor_load_to_lds(g0, g1, z4, z4, z8, 0);
#else
  __builtin_amdgcn_tensor_load_to_lds(g0, g1, z4, z4, 0);
#endif
}

// ---------------------------------------------------------------------------
// Setup: w[n,k] = U[n,:].V[idx[n,k],:]; scatter into dense transposed weight
// WdT[n,m] (bf16, n-major so TDM B-tiles are row-contiguous). Duplicate idx
// entries accumulate via LDS f32 atomics.
// ---------------------------------------------------------------------------
__global__ __launch_bounds__(256) void snn_build_wdt(
    const float* __restrict__ U, const float* __restrict__ V,
    const int* __restrict__ idx, bf16_t* __restrict__ WdT) {
  __shared__ float row[NN];                       // 16 KB LDS
  const int n = blockIdx.x;
  for (int i = threadIdx.x; i < NN; i += 256) row[i] = 0.0f;
  __syncthreads();
  if (threadIdx.x < KK) {
    const int m = idx[n * KK + threadIdx.x];
    const float* u  = U + (size_t)n * RR;
    const float* vv = V + (size_t)m * RR;
    float wv = 0.0f;
#pragma unroll
    for (int r = 0; r < RR; ++r) wv += u[r] * vv[r];
    atomicAdd(&row[m], wv);                       // ds_add_f32
  }
  __syncthreads();
  bf16_t* out = WdT + (size_t)n * NN;
  for (int i = threadIdx.x; i < NN; i += 256) out[i] = (bf16_t)row[i];
}

__global__ void snn_init(float* vbuf, bf16_t* acc0, bf16_t* acc1, float* spk) {
  const int i = blockIdx.x * blockDim.x + threadIdx.x;
  if (i < BB * NN) {
    vbuf[i] = 0.0f;
    acc0[i] = (bf16_t)0.0f;
    acc1[i] = (bf16_t)0.0f;
  }
  if (i == 0) *spk = 0.0f;
}

// ---------------------------------------------------------------------------
// One timestep. fb = accIn x WdT^T via WMMA, operands staged by the TDM into
// double-buffered LDS tiles; fused LIF epilogue; acc double-buffered across
// launches for the grid-wide recurrence dependency.
// Block: 256 thr = 8 waves, owns 128 output columns x all 32 batch rows.
// ---------------------------------------------------------------------------
__global__ __launch_bounds__(256) void snn_step(
    const float* __restrict__ x, const bf16_t* __restrict__ WdT,
    const bf16_t* __restrict__ accIn, bf16_t* __restrict__ accOut,
    float* __restrict__ vbuf, float* __restrict__ spkTotal, int t) {
  __shared__ __align__(64) bf16_t lA[2][BB * KC];    //  8 KB: acc rows tile
  __shared__ __align__(64) bf16_t lB[2][128 * KC];   // 32 KB: WdT cols tile
  __shared__ float red[256];

  const int tid  = threadIdx.x;
  const int lane = tid & 31;
  const int wave = tid >> 5;
  const int cl   = lane & 15;
  const int kh   = lane >> 4;
  const int col  = blockIdx.x * 128 + wave * 16 + cl;

  v8f c[2] = {};                      // f32 accumulators: b 0-15 / b 16-31

  if (t > 0) {
    const bf16_t* gA = accIn;
    const bf16_t* gB = WdT + (size_t)(blockIdx.x * 128) * NN;
    if (wave == 0) {                  // single wave drives the TDM
      tdm_load_tile_2d(gA, &lA[0][0], BB);
      tdm_load_tile_2d(gB, &lB[0][0], 128);
    }
    for (int ch = 0; ch < NCH; ++ch) {
      const int buf = ch & 1;
      if (wave == 0) {
        if (ch + 1 < NCH) {           // prefetch chunk ch+1 into other buffer
          tdm_load_tile_2d(gA + (ch + 1) * KC, &lA[buf ^ 1][0], BB);
          tdm_load_tile_2d(gB + (ch + 1) * KC, &lB[buf ^ 1][0], 128);
          __builtin_amdgcn_s_wait_tensorcnt(2);   // chunk ch resident
        } else {
          __builtin_amdgcn_s_wait_tensorcnt(0);
        }
      }
      __syncthreads();                // publish chunk ch to all waves

      const bf16_t* la0 = &lA[buf][cl * KC];
      const bf16_t* la1 = &lA[buf][(cl + 16) * KC];
      const bf16_t* lb  = &lB[buf][(wave * 16 + cl) * KC];
      const int aoff = kh * 16;       // contiguous k order: lanes 16-31 = K+16
      const int boff = kh * 16;       // contiguous k order: lanes 16-31 = K+16
#pragma unroll
      for (int ks = 0; ks < KC; ks += 32) {
        ALoad a0, a1;
        a0.u[0] = *(const uint4*)(la0 + ks + aoff);       // ds_load_b128
        a0.u[1] = *(const uint4*)(la0 + ks + aoff + 8);
        a1.u[0] = *(const uint4*)(la1 + ks + aoff);
        a1.u[1] = *(const uint4*)(la1 + ks + aoff + 8);
        v16bf bm = *(const v16bf*)(lb + ks + boff);
        c[0] = __builtin_amdgcn_wmma_f32_16x16x32_bf16(
            false, a0.v, false, bm, (short)0, c[0], false, false);
        c[1] = __builtin_amdgcn_wmma_f32_16x16x32_bf16(
            false, a1.v, false, bm, (short)0, c[1], false, false);
      }
      __syncthreads();                // done consuming buf: safe to overwrite
    }
  }

  // Fused LIF epilogue. C layout: c[r] <-> (M = kh*8 + r, N = cl).
  const float inv_t1 = 1.0f / (float)(t + 1);
  float spk = 0.0f;
#pragma unroll
  for (int half = 0; half < 2; ++half) {
#pragma unroll
    for (int r = 0; r < 8; ++r) {
      const int b = half * 16 + kh * 8 + r;
      const size_t off = (size_t)b * NN + col;
      const float fb = (t > 0) ? c[half][r] * inv_t1 : 0.0f;
      const float xv = x[((size_t)b * TT + t) * NN + col];
      float vm = ALPHA * vbuf[off] + xv + fb;
      const float s = (vm >= V_TH) ? 1.0f : 0.0f;
      vbuf[off] = vm * (1.0f - s);
      accOut[off] = (bf16_t)((float)accIn[off] + s);  // exact: integer <= 256
      spk += s;
    }
  }

  red[tid] = spk;
  __syncthreads();
  for (int st = 128; st > 0; st >>= 1) {
    if (tid < st) red[tid] += red[tid + st];
    __syncthreads();
  }
  if (tid == 0) atomicAdd(spkTotal, red[0]);
}

__global__ void snn_finalize(const bf16_t* __restrict__ accF,
                             const float* __restrict__ spkTotal,
                             float* __restrict__ out) {
  const int i = blockIdx.x * blockDim.x + threadIdx.x;
  if (i < BB * NN) out[i] = (float)accF[i];
  if (i == 0) out[BB * NN] = *spkTotal / (float)((size_t)BB * NN * TT);
}

extern "C" void kernel_launch(void* const* d_in, const int* in_sizes, int n_in,
                              void* d_out, int out_size, void* d_ws, size_t ws_size,
                              hipStream_t stream) {
  const float* x  = (const float*)d_in[0];   // [B,T,N]
  const float* U  = (const float*)d_in[1];   // [N,R]
  const float* V  = (const float*)d_in[2];   // [N,R]
  const int*  idx = (const int*)d_in[3];     // [N,K]
  float* out = (float*)d_out;

  char* ws = (char*)d_ws;
  bf16_t* WdT = (bf16_t*)ws;                                   // 32 MB, L2-resident
  size_t o = (size_t)NN * NN * sizeof(bf16_t);
  float*  vbuf = (float*)(ws + o);  o += (size_t)BB * NN * sizeof(float);
  bf16_t* acc0 = (bf16_t*)(ws + o); o += (size_t)BB * NN * sizeof(bf16_t);
  bf16_t* acc1 = (bf16_t*)(ws + o); o += (size_t)BB * NN * sizeof(bf16_t);
  float*  spkTot = (float*)(ws + o);

  snn_build_wdt<<<NN, 256, 0, stream>>>(U, V, idx, WdT);
  snn_init<<<(BB * NN + 255) / 256, 256, 0, stream>>>(vbuf, acc0, acc1, spkTot);

  bf16_t* bufs[2] = {acc0, acc1};
  for (int t = 0; t < TT; ++t) {
    snn_step<<<NN / 128, 256, 0, stream>>>(x, WdT, bufs[t & 1],
                                           bufs[(t + 1) & 1], vbuf, spkTot, t);
  }
  snn_finalize<<<(BB * NN + 511) / 256, 256, 0, stream>>>(bufs[TT & 1], spkTot, out);
}
